// GPT_3341484556647
// MI455X (gfx1250) — compile-verified, matrix operand order checked by reading, not run on hardware
//
#include <hip/hip_runtime.h>
#include <hip/hip_bf16.h>

// ---------------------------------------------------------------------------
// GPT forward for MI455X (gfx1250): bf16 WMMA GEMMs, TDM (tensor_load_to_lds)
// double-buffered LDS staging, all B operands pre-transposed to [N,K] so every
// tile is K-contiguous (uniform ds_load_2addr_b64 fragment reads).
// ---------------------------------------------------------------------------

typedef __attribute__((ext_vector_type(16))) __bf16 v16bf;
typedef __attribute__((ext_vector_type(8)))  float  v8f;
typedef __attribute__((ext_vector_type(2)))  __bf16 bf16x2;
typedef __attribute__((ext_vector_type(4)))  unsigned int v4u;
typedef __attribute__((ext_vector_type(8)))  int v8i;
typedef __attribute__((ext_vector_type(4)))  int v4i;

// Model dims (match reference)
constexpr int Bz = 8, Tt = 512, Cc = 1024, Hh = 16, Ll = 8, Vv = 32000;
constexpr int Dd = Cc / Hh, FFf = 4 * Cc;

// Epilogue flags
constexpr int EPI_BIAS = 1, EPI_GELU = 2, EPI_RES = 4, EPI_F32 = 8,
              EPI_BF16 = 16, EPI_MASKSCALE = 32, EPI_VSTORET = 64;

// ---------------------------------------------------------------------------
// TDM: 2-D tile load global->LDS via Tensor Data Mover (CDNA5 ISA ch.8 D#).
//   tile_d0   : tile width in (1<<ds_log2)-byte units (memory-contiguous dim)
//   tile_d1   : number of rows
//   stride_u  : row stride in (1<<ds_log2)-byte units
//   pad_int   : LDS pad after 8*2^pad_int bytes; pad_amt: (pad_amt+1) DWORDs
// ---------------------------------------------------------------------------
static __device__ __forceinline__ void tdm_load_2d(
    unsigned lds_addr, const void* gptr, unsigned tile_d0, unsigned tile_d1,
    unsigned long long stride_u, unsigned ds_log2, unsigned pad_int,
    unsigned pad_amt) {
  const unsigned long long ga = (unsigned long long)gptr;
  v4u g0;
  g0[0] = 1u;                                   // count=1, no gather
  g0[1] = lds_addr;                             // lds_addr [63:32]
  g0[2] = (unsigned)(ga & 0xffffffffu);         // global_addr lo
  g0[3] = (unsigned)((ga >> 32) & 0x1ffffffu)   // global_addr [56:32]
          | (2u << 30);                         // type=2 ("image")
  v8i g1;
  g1[0] = (int)((ds_log2 << 16) | (1u << 20)    // data_size | pad_enable
                | (pad_int << 22) | (pad_amt << 25));
  g1[1] = (int)((tile_d0 & 0xffffu) << 16);               // tensor_dim0 lo
  g1[2] = (int)((tile_d0 >> 16) | ((tile_d1 & 0xffffu) << 16)); // td0 hi|td1 lo
  g1[3] = (int)((tile_d1 >> 16) | ((tile_d0 & 0xffffu) << 16)); // td1 hi|tile_dim0
  g1[4] = (int)(tile_d1 & 0xffffu);                       // tile_dim1, tile_dim2=0
  g1[5] = (int)(unsigned)(stride_u & 0xffffffffull);      // dim0_stride lo
  g1[6] = (int)(unsigned)((stride_u >> 32) & 0xffffull);  // dim0_stride hi
  g1[7] = 0;                                              // dim1_stride (2-D)
  const v4i z4 = {0, 0, 0, 0};
#if __clang_major__ >= 23
  const v8i z8 = {0, 0, 0, 0, 0, 0, 0, 0};
  __builtin_amdgcn_tensor_load_to_lds(g0, g1, z4, z4, z8, 0);
#else
  __builtin_amdgcn_tensor_load_to_lds(g0, g1, z4, z4, 0);
#endif
}

// ---------------------------------------------------------------------------
// Transpose + fp32->bf16: dst[N,K] = (bf16) src[K,N], batched over blockIdx.z.
// 32x32 tiles staged through LDS; both global accesses coalesced.
// ---------------------------------------------------------------------------
__global__ __launch_bounds__(256)
void cvtT_kernel(const float* __restrict__ src, __bf16* __restrict__ dst,
                 int K, int N) {
  __shared__ float tile[32][33];
  const long long zoff = (long long)blockIdx.z * K * N;
  const int k0 = blockIdx.y * 32, n0 = blockIdx.x * 32;
  const int tx = threadIdx.x & 31, ty = threadIdx.x >> 5;  // 32 x 8
#pragma unroll
  for (int i = ty; i < 32; i += 8)
    tile[i][tx] = src[zoff + (long long)(k0 + i) * N + n0 + tx];
  __syncthreads();
#pragma unroll
  for (int i = ty; i < 32; i += 8)
    dst[zoff + (long long)(n0 + i) * K + k0 + tx] = (__bf16)tile[tx][i];
}

// ---------------------------------------------------------------------------
// Embedding: x[b,t,:] = tok_emb[idx[b,t]] + pos_emb[t] + type_emb[1]
// ---------------------------------------------------------------------------
__global__ __launch_bounds__(256)
void embed_kernel(const int* __restrict__ idx, const float* __restrict__ tok,
                  const float* __restrict__ type_emb,
                  const float* __restrict__ pos, float* __restrict__ x) {
  const int row = blockIdx.x;            // b*T + t
  const int t = row % Tt;
  const long long tokid = idx[row];
#pragma unroll
  for (int i = 0; i < 4; ++i) {
    int c = threadIdx.x + i * 256;
    x[(long long)row * Cc + c] =
        tok[tokid * Cc + c] + pos[(long long)t * Cc + c] + type_emb[Cc + c];
  }
}

// ---------------------------------------------------------------------------
// LayerNorm over C=1024, one row per 256-thread block, bf16 output
// ---------------------------------------------------------------------------
__global__ __launch_bounds__(256)
void ln_kernel(const float* __restrict__ x, const float* __restrict__ g,
               const float* __restrict__ b, __bf16* __restrict__ out) {
  const int row = blockIdx.x;
  const int tid = threadIdx.x;
  const float* xr = x + (long long)row * Cc;
  __shared__ float red[8];

  float v0[4], s = 0.f;
#pragma unroll
  for (int i = 0; i < 4; ++i) { v0[i] = xr[tid + i * 256]; s += v0[i]; }
#pragma unroll
  for (int off = 16; off > 0; off >>= 1) s += __shfl_xor(s, off, 32);
  if ((tid & 31) == 0) red[tid >> 5] = s;
  __syncthreads();
  float mean = 0.f;
#pragma unroll
  for (int i = 0; i < 8; ++i) mean += red[i];
  mean *= (1.0f / Cc);
  __syncthreads();

  float vs = 0.f;
#pragma unroll
  for (int i = 0; i < 4; ++i) { float d = v0[i] - mean; vs += d * d; }
#pragma unroll
  for (int off = 16; off > 0; off >>= 1) vs += __shfl_xor(vs, off, 32);
  if ((tid & 31) == 0) red[tid >> 5] = vs;
  __syncthreads();
  float var = 0.f;
#pragma unroll
  for (int i = 0; i < 8; ++i) var += red[i];
  var *= (1.0f / Cc);
  const float rstd = rsqrtf(var + 1e-5f);

#pragma unroll
  for (int i = 0; i < 4; ++i) {
    int c = tid + i * 256;
    out[(long long)row * Cc + c] =
        (__bf16)((v0[i] - mean) * rstd * g[c] + b[c]);
  }
}

// ---------------------------------------------------------------------------
// Row softmax over T=512 (scores pre-scaled & causally masked to -1e30).
// One wave32 per row, 8 rows per block. In-place on bf16.
// ---------------------------------------------------------------------------
__global__ __launch_bounds__(256)
void softmax_kernel(__bf16* __restrict__ att) {
  const int row = blockIdx.x * 8 + (threadIdx.x >> 5);
  const int lane = threadIdx.x & 31;
  __bf16* ar = att + (long long)row * Tt;

  float v[16], mx = -3.0e38f;
#pragma unroll
  for (int i = 0; i < 16; ++i) {
    v[i] = (float)ar[lane + i * 32];
    mx = fmaxf(mx, v[i]);
  }
#pragma unroll
  for (int off = 16; off > 0; off >>= 1) mx = fmaxf(mx, __shfl_xor(mx, off, 32));
  float s = 0.f;
#pragma unroll
  for (int i = 0; i < 16; ++i) { v[i] = __expf(v[i] - mx); s += v[i]; }
#pragma unroll
  for (int off = 16; off > 0; off >>= 1) s += __shfl_xor(s, off, 32);
  const float inv = 1.0f / s;
#pragma unroll
  for (int i = 0; i < 16; ++i) ar[lane + i * 32] = (__bf16)(v[i] * inv);
}

// ---------------------------------------------------------------------------
// Batched bf16 WMMA GEMM, TDM double-buffered LDS staging.
//   A : [M,K] row-major bf16.
//   Bt: B TRANSPOSED, i.e. [N,K] row-major (B[k][n] = Bt[n*ldb + k]).
//   Per-batch offsets: (z/div)*so + (z%div)*si for A, B, C.
//   BK=32 = one v_wmma_f32_16x16x32_bf16 K-step per stage.
//   LDS: both tiles K-contiguous rows of 64B + 8B TDM-written pad.
// ---------------------------------------------------------------------------
template <int BM, int BN, int WAVES_M, int WAVES_N, int EPI>
__global__ __launch_bounds__(WAVES_M* WAVES_N * 32)
void gemm_wmma(const __bf16* __restrict__ Ag, int lda, long long sAo,
               long long sAi, int divA, const __bf16* __restrict__ Btg,
               int ldb, long long sBo, long long sBi, int divB,
               float* __restrict__ Cf, __bf16* __restrict__ Cb, int ldc,
               long long sCo, long long sCi, int divC,
               const float* __restrict__ bias, const float* __restrict__ resid,
               float scale, int K) {
  constexpr int BK = 32;
  constexpr int LW = BK + 4;                       // LDS row in bf16 (72B)
  constexpr int WM = BM / WAVES_M, WN = BN / WAVES_N;
  constexpr int TM = WM / 16, TN = WN / 16;

  __shared__ __align__(16) __bf16 As[2][BM * LW];
  __shared__ __align__(16) __bf16 Bs[2][BN * LW];

  const int tid = threadIdx.x;
  const int lane = tid & 31;
  const int wave = tid >> 5;
  const int wm = wave % WAVES_M;
  const int wn = wave / WAVES_M;
  const int z = blockIdx.z;

  const __bf16* Ab = Ag + (long long)(z / divA) * sAo + (long long)(z % divA) * sAi;
  const __bf16* Bb = Btg + (long long)(z / divB) * sBo + (long long)(z % divB) * sBi;
  const long long coff = (long long)(z / divC) * sCo + (long long)(z % divC) * sCi;

  const int mBase = blockIdx.y * BM;
  const int nBase = blockIdx.x * BN;

  v8f acc[TM][TN];
  const v8f zf = {0.f, 0.f, 0.f, 0.f, 0.f, 0.f, 0.f, 0.f};
#pragma unroll
  for (int i = 0; i < TM; ++i)
#pragma unroll
    for (int j = 0; j < TN; ++j) acc[i][j] = zf;

  const int mlane = lane & 15;   // row (A) / col (B,C) within 16-tile
  const int khalf = lane >> 4;   // K-half select (A/B), +8 rows (C/D)

  // Issue both TDM tile loads for K-step k0 into buffer `buf`.
  auto issue_tiles = [&](int k0, int buf) {
    tdm_load_2d((unsigned)(unsigned long long)(void*)&As[buf][0],
                Ab + (long long)mBase * lda + k0,
                /*tile_d0=*/BK * 2 / 8, /*tile_d1=*/BM,
                (unsigned long long)lda * 2 / 8, /*ds_log2=*/3,
                /*pad_int=*/3, /*pad_amt=*/1);              // 64B rows + 8B pad
    tdm_load_2d((unsigned)(unsigned long long)(void*)&Bs[buf][0],
                Bb + (long long)nBase * ldb + k0, BK * 2 / 8, BN,
                (unsigned long long)ldb * 2 / 8, 3, 3, 1);  // 64B rows + 8B pad
  };

  // Prologue: start DMA of first tile pair.
  if (wave == 0) issue_tiles(0, 0);

  int cur = 0;
  for (int k0 = 0; k0 < K; k0 += BK) {
    if (wave == 0) {
      if (k0 + BK < K) {
        issue_tiles(k0 + BK, cur ^ 1);          // overlap next DMA w/ compute
        __builtin_amdgcn_s_wait_tensorcnt(2);   // in-order => current pair done
      } else {
        __builtin_amdgcn_s_wait_tensorcnt(0);
      }
    }
    __syncthreads();

    // ---- fragments (ISA 7.12.2 wave32 16-bit layouts) ----
    v16bf afr[TM], bfr[TN];
#pragma unroll
    for (int i = 0; i < TM; ++i) {
      bf16x2* ap = (bf16x2*)&afr[i];
      const int row = wm * WM + i * 16 + mlane;
#pragma unroll
      for (int v = 0; v < 8; ++v) {
        const int kk = ((v & 4) ? 16 : 0) + khalf * 8 + 2 * (v & 3);
        ap[v] = *(const bf16x2*)&As[cur][row * LW + kk];
      }
    }
#pragma unroll
    for (int j = 0; j < TN; ++j) {
      bf16x2* bp = (bf16x2*)&bfr[j];
      const int col = wn * WN + j * 16 + mlane;
#pragma unroll
      for (int v = 0; v < 8; ++v) {
        const int kk = ((v & 4) ? 16 : 0) + khalf * 8 + 2 * (v & 3);
        bp[v] = *(const bf16x2*)&Bs[cur][col * LW + kk];
      }
    }
#pragma unroll
    for (int i = 0; i < TM; ++i)
#pragma unroll
      for (int j = 0; j < TN; ++j)
        acc[i][j] = __builtin_amdgcn_wmma_f32_16x16x32_bf16(
            false, afr[i], false, bfr[j], (short)0, acc[i][j], false, false);
    __syncthreads();   // all reads of As/Bs[cur] done before TDM reuses it
    cur ^= 1;
  }

  // ---- epilogue (C/D layout: vgpr e -> row e + 8*khalf, lane -> col) ----
#pragma unroll
  for (int i = 0; i < TM; ++i) {
#pragma unroll
    for (int j = 0; j < TN; ++j) {
      const int col = nBase + wn * WN + j * 16 + mlane;
#pragma unroll
      for (int e = 0; e < 8; ++e) {
        const int row = mBase + wm * WM + i * 16 + e + khalf * 8;
        float v = acc[i][j][e];
        if (EPI & EPI_BIAS) v += bias[col];
        if (EPI & EPI_MASKSCALE) v = (col <= row) ? v * scale : -1e30f;
        if (EPI & EPI_GELU) v = 0.5f * v * (1.0f + erff(v * 0.70710678118654752f));
        const long long oix = coff + (long long)row * ldc + col;
        if (EPI & EPI_RES) v += resid[oix];
        if (EPI & EPI_F32) Cf[oix] = v;
        if (EPI & EPI_BF16) Cb[oix] = (__bf16)v;
        if (EPI & EPI_VSTORET) {  // store transposed: vT[b][col][t], t=row%T
          const long long tix = (long long)(row >> 9) * ((long long)Cc * Tt) +
                                (long long)col * Tt + (row & (Tt - 1));
          Cb[tix] = (__bf16)v;
        }
      }
    }
  }
}

// ---------------------------------------------------------------------------
// Host orchestration
// ---------------------------------------------------------------------------
extern "C" void kernel_launch(void* const* d_in, const int* in_sizes, int n_in,
                              void* d_out, int out_size, void* d_ws,
                              size_t ws_size, hipStream_t stream) {
  (void)in_sizes; (void)n_in; (void)out_size; (void)ws_size;

  const int*   idx   = (const int*)d_in[0];
  const float* tok   = (const float*)d_in[1];
  const float* typeE = (const float*)d_in[2];
  const float* pos   = (const float*)d_in[3];
  const float* ln1g  = (const float*)d_in[4];
  const float* ln1b  = (const float*)d_in[5];
  const float* Wq    = (const float*)d_in[6];
  const float* bq    = (const float*)d_in[7];
  const float* Wk    = (const float*)d_in[8];
  const float* bk    = (const float*)d_in[9];
  const float* Wv    = (const float*)d_in[10];
  const float* bv    = (const float*)d_in[11];
  const float* Wo    = (const float*)d_in[12];
  const float* bo    = (const float*)d_in[13];
  const float* ln2g  = (const float*)d_in[14];
  const float* ln2b  = (const float*)d_in[15];
  const float* W1    = (const float*)d_in[16];
  const float* b1    = (const float*)d_in[17];
  const float* W2    = (const float*)d_in[18];
  const float* b2    = (const float*)d_in[19];
  const float* lnfg  = (const float*)d_in[20];
  const float* lnfb  = (const float*)d_in[21];
  const float* Whead = (const float*)d_in[22];
  float* out = (float*)d_out;

  // ---- workspace carve-up (256B aligned) ----
  char* p = (char*)d_ws;
  auto take = [&](size_t elems, size_t elsz) -> void* {
    void* r = (void*)p;
    p += (elems * elsz + 255) & ~(size_t)255;
    return r;
  };
  const size_t szW  = (size_t)Ll * Cc * Cc;
  const size_t szW1 = (size_t)Ll * Cc * FFf;
  const size_t szWh = (size_t)Cc * Vv;
  const size_t szX  = (size_t)Bz * Tt * Cc;
  const size_t szA  = (size_t)Bz * Hh * Tt * Tt;
  const size_t szF  = (size_t)Bz * Tt * FFf;

  // All weights stored TRANSPOSED ([N,K] row-major) in bf16.
  __bf16* wqT = (__bf16*)take(szW, 2);
  __bf16* wkT = (__bf16*)take(szW, 2);
  __bf16* wvT = (__bf16*)take(szW, 2);
  __bf16* woT = (__bf16*)take(szW, 2);
  __bf16* w1T = (__bf16*)take(szW1, 2);
  __bf16* w2T = (__bf16*)take(szW1, 2);
  __bf16* whT = (__bf16*)take(szWh, 2);
  float*  xf  = (float*)take(szX, 4);
  __bf16* h_b = (__bf16*)take(szX, 2);
  __bf16* q_b = (__bf16*)take(szX, 2);
  __bf16* k_b = (__bf16*)take(szX, 2);
  __bf16* vT  = (__bf16*)take(szX, 2);   // [B][C][T] (per-head transposed V)
  __bf16* y_b = (__bf16*)take(szX, 2);
  __bf16* att_b = (__bf16*)take(szA, 2);
  __bf16* f1_b  = (__bf16*)take(szF, 2);

  // ---- transpose + convert weights to bf16 ----
  const dim3 tcw(Cc / 32, Cc / 32, Ll);     // per-layer [C,C] -> [C,C]^T
  cvtT_kernel<<<tcw, 256, 0, stream>>>(Wq, wqT, Cc, Cc);
  cvtT_kernel<<<tcw, 256, 0, stream>>>(Wk, wkT, Cc, Cc);
  cvtT_kernel<<<tcw, 256, 0, stream>>>(Wv, wvT, Cc, Cc);
  cvtT_kernel<<<tcw, 256, 0, stream>>>(Wo, woT, Cc, Cc);
  cvtT_kernel<<<dim3(FFf / 32, Cc / 32, Ll), 256, 0, stream>>>(W1, w1T, Cc, FFf);
  cvtT_kernel<<<dim3(Cc / 32, FFf / 32, Ll), 256, 0, stream>>>(W2, w2T, FFf, Cc);
  cvtT_kernel<<<dim3(Vv / 32, Cc / 32, 1), 256, 0, stream>>>(Whead, whT, Cc, Vv);

  // ---- embeddings ----
  embed_kernel<<<Bz * Tt, 256, 0, stream>>>(idx, tok, typeE, pos, xf);

  const float scaleAtt = 0.125f;  // 1/sqrt(D)
  const dim3 gProj(Cc / 128, (Bz * Tt) / 128, 1);       // N=1024
  const dim3 gFF1(FFf / 128, (Bz * Tt) / 128, 1);       // N=4096
  const dim3 gScore(Tt / 128, Tt / 128, Bz * Hh);       // 512x512 x128 batches
  const dim3 gY(1, Tt / 128, Bz * Hh);                  // N=64
  const dim3 gHead(Vv / 128, (Bz * Tt) / 128, 1);       // N=32000

  for (int l = 0; l < Ll; ++l) {
    const __bf16* wq = wqT + (size_t)l * Cc * Cc;
    const __bf16* wk = wkT + (size_t)l * Cc * Cc;
    const __bf16* wv = wvT + (size_t)l * Cc * Cc;
    const __bf16* wo = woT + (size_t)l * Cc * Cc;
    const __bf16* w1 = w1T + (size_t)l * Cc * FFf;
    const __bf16* w2 = w2T + (size_t)l * FFf * Cc;

    // h = LN1(x)
    ln_kernel<<<Bz * Tt, 256, 0, stream>>>(xf, ln1g + l * Cc, ln1b + l * Cc, h_b);

    // q,k = h @ Wq/Wk + b (bf16); v stored transposed vT[b][c][t]
    gemm_wmma<128, 128, 2, 4, EPI_BIAS | EPI_BF16>
        <<<gProj, 256, 0, stream>>>(h_b, Cc, 0, 0, 1, wq, Cc, 0, 0, 1, nullptr,
                                    q_b, Cc, 0, 0, 1, bq + l * Cc, nullptr, 1.f, Cc);
    gemm_wmma<128, 128, 2, 4, EPI_BIAS | EPI_BF16>
        <<<gProj, 256, 0, stream>>>(h_b, Cc, 0, 0, 1, wk, Cc, 0, 0, 1, nullptr,
                                    k_b, Cc, 0, 0, 1, bk + l * Cc, nullptr, 1.f, Cc);
    gemm_wmma<128, 128, 2, 4, EPI_BIAS | EPI_VSTORET>
        <<<gProj, 256, 0, stream>>>(h_b, Cc, 0, 0, 1, wv, Cc, 0, 0, 1, nullptr,
                                    vT, Cc, 0, 0, 1, bv + l * Cc, nullptr, 1.f, Cc);

    // scores = scale * q @ k^T, causal masked, bf16  (batch z = b*H + h)
    gemm_wmma<128, 128, 2, 4, EPI_MASKSCALE | EPI_BF16>
        <<<gScore, 256, 0, stream>>>(q_b, Cc, (long long)Tt * Cc, Dd, Hh,
                                     k_b, Cc, (long long)Tt * Cc, Dd, Hh,
                                     nullptr, att_b, Tt, (long long)Tt * Tt, 0, 1,
                                     nullptr, nullptr, scaleAtt, Dd);

    // softmax rows (in place)
    softmax_kernel<<<(Bz * Hh * Tt) / 8, 256, 0, stream>>>(att_b);

    // y = att @ v : B^T = vT slice [D rows of T], ldb=T  (M=512,N=64,K=512)
    gemm_wmma<128, 64, 4, 2, EPI_BF16>
        <<<gY, 256, 0, stream>>>(att_b, Tt, (long long)Tt * Tt, 0, 1,
                                 vT, Tt, (long long)Cc * Tt, (long long)Dd * Tt, Hh,
                                 nullptr, y_b, Cc, (long long)Tt * Cc, Dd, Hh,
                                 nullptr, nullptr, 1.f, Tt);

    // x = x + y @ Wo + bo   (fp32 out, residual)
    gemm_wmma<128, 128, 2, 4, EPI_BIAS | EPI_RES | EPI_F32>
        <<<gProj, 256, 0, stream>>>(y_b, Cc, 0, 0, 1, wo, Cc, 0, 0, 1, xf,
                                    nullptr, Cc, 0, 0, 1, bo + l * Cc, xf, 1.f, Cc);

    // h = LN2(x)
    ln_kernel<<<Bz * Tt, 256, 0, stream>>>(xf, ln2g + l * Cc, ln2b + l * Cc, h_b);

    // f1 = gelu(h @ W1 + b1)  (bf16)
    gemm_wmma<128, 128, 2, 4, EPI_BIAS | EPI_GELU | EPI_BF16>
        <<<gFF1, 256, 0, stream>>>(h_b, Cc, 0, 0, 1, w1, Cc, 0, 0, 1, nullptr,
                                   f1_b, FFf, 0, 0, 1, b1 + l * FFf, nullptr, 1.f, Cc);

    // x = x + f1 @ W2 + b2
    gemm_wmma<128, 128, 2, 4, EPI_BIAS | EPI_RES | EPI_F32>
        <<<gProj, 256, 0, stream>>>(f1_b, FFf, 0, 0, 1, w2, FFf, 0, 0, 1, xf,
                                    nullptr, Cc, 0, 0, 1, b2 + l * Cc, xf, 1.f, FFf);
  }

  // final LN + vocab head (fp32 logits straight to d_out)
  ln_kernel<<<Bz * Tt, 256, 0, stream>>>(xf, lnfg, lnfb, h_b);
  gemm_wmma<128, 128, 2, 4, EPI_F32>
      <<<gHead, 256, 0, stream>>>(h_b, Cc, 0, 0, 1, whT, Cc, 0, 0, 1, out,
                                  nullptr, Vv, 0, 0, 1, nullptr, nullptr, 1.f, Cc);
}
